// convNd_7524782703129
// MI455X (gfx1250) — matrix operand order repaired and checked by reading, not compiled, not run on hardware
//
#include <hip/hip_runtime.h>
#include <stdint.h>

// ---- WMMA operand/accumulator vector types (CDNA5 gfx1250, wave32) ----
typedef __attribute__((ext_vector_type(16))) __bf16 v16bf;
typedef __attribute__((ext_vector_type(8)))  float  v8f;

#define CIN     16
#define COUT    32
#define DDIM    24
#define S1      24          // flat stride of d2 (one d3 row)
#define S2      576         // flat stride of d1
#define S3      13824       // flat stride of d0
#define SPATIAL 331776      // 24^4
#define BATCH   2
#define MTOT    (BATCH*SPATIAL)
#define TAPS    81
#define BM      256         // M rows per workgroup
#define WIN     306         // window positions: 256 rows + 25 halo each side
#define AROW    24          // halfwords per window row (16 data + 8 pad -> 48B)
#define ZROW    306         // guaranteed-zero window row

// ws-resident B matrix, pre-expanded to per-lane WMMA fragment order:
//   [ij 0..8][q 0..4][frag 0..1][lane 0..31][c 0..15]  (bf16)
// element-chunk tap selection, lane-half channel offset and the q==4 zero-pad tap are baked in at prep time.
#define WB_ELEMS (9*5*2*32*16)     // 46080 bf16 = 92160 B

__device__ __forceinline__ unsigned short f2bf(float f) {
    union { float f; unsigned u; } x; x.f = f;
    unsigned u = x.u;
    unsigned r = u + 0x7FFFu + ((u >> 16) & 1u);   // round-to-nearest-even
    return (unsigned short)(r >> 16);
}

// ---- prep: W (f32, [n][c][tap]) -> ws bf16 in exact fragment-lane order ----
__global__ void prep_w_kernel(const float* __restrict__ Wt,
                              unsigned short* __restrict__ wb) {
    int e = blockIdx.x * 256 + threadIdx.x;
    if (e >= WB_ELEMS) return;
    int c    = e & 15;
    int lane = (e >> 4) & 31;
    int frag = (e >> 9) & 1;
    int qij  = e >> 10;
    int q    = qij % 5;
    int ij   = qij / 5;
    int tloc = 2 * q + (c >> 3);           // element chunk selects tap t0/t1
    int n    = frag * 16 + (lane & 15);    // fragment column
    unsigned short v = 0;
    if (tloc < 9)                          // tloc==9 (q==4, elements 8..15) -> K pad zeros
        v = f2bf(Wt[(n * CIN + (8 * (lane >> 4) + (c & 7))) * TAPS + (ij * 9 + tloc)]);
    wb[e] = v;
}

__launch_bounds__(256, 2)
__global__ void conv4d_wmma_kernel(const float* __restrict__ x,
                                   const unsigned short* __restrict__ wb,
                                   float* __restrict__ out) {
    // x window for current (i,j): [pos 0..305][c 0..15]; rows 306/307 stay zero
    __shared__ __align__(16) unsigned short Awin[308 * AROW];   // 14784 B

    const int tid  = threadIdx.x;
    const int lane = tid & 31;
    const int half = lane >> 4;
    const int l15  = lane & 15;
    const int wv   = tid >> 5;                 // wave 0..7 -> 32-row M subtile

    const int tile_m = blockIdx.x * BM;
    const int b      = tile_m / SPATIAL;       // constant per block (SPATIAL % BM == 0)
    const int tile_s = tile_m - b * SPATIAL;

    // ---- per-lane fragment-row constants (two 16-row groups per wave) ----
    const int r0    = wv * 32 + l15;
    const int s0    = tile_s + r0;
    const int s1    = s0 + 16;
    const int d2r0  = (s0 / S1) % DDIM,  d3r0 = s0 % S1;
    const int d2r1  = (s1 / S1) % DDIM,  d3r1 = s1 % S1;
    const unsigned short* arow0 = &Awin[(r0 + 25) * AROW + 8 * half];
    const unsigned short* arow1 = arow0 + 16 * AROW;
    const unsigned short* azero = &Awin[ZROW * AROW + 8 * half];

    // ---- single per-lane B base; all step offsets become immediates ----
    const unsigned short* blane = wb + lane * 16;   // 32 B per lane per fragment

    // ---- staging roles: 16 lanes load 16 consecutive positions of one channel ----
    const int st_c  = tid >> 4;                // channel 0..15
    const int st_p0 = tid & 15;                // starting window position

    v8f acc00 = {0.f,0.f,0.f,0.f,0.f,0.f,0.f,0.f};
    v8f acc01 = acc00, acc10 = acc00, acc11 = acc00;

    for (int ij = 0; ij < 9; ++ij) {
        const int di = ij / 3 - 1;
        const int dj = ij % 3 - 1;

        __syncthreads();   // previous group's readers done

        // ---------- stage x window (d0/d1 shift + validity folded in) ----------
        {
            const int shift = di * S3 + dj * S2;
            const float* xc = x + (size_t)(b * CIN + st_c) * SPATIAL;
            for (int p = st_p0; p < 308; p += 16) {
                unsigned short v = 0;
                if (p < WIN) {
                    int sp = tile_s - 25 + p;                  // unshifted flat pos
                    if ((unsigned)sp < SPATIAL) {
                        int d0 = sp / S3;
                        int d1 = (sp / S2) % DDIM;
                        if (((unsigned)(d0 + di) < DDIM) & ((unsigned)(d1 + dj) < DDIM))
                            v = f2bf(xc[sp + shift]);
                    }
                }
                Awin[p * AROW + st_c] = v;                     // rows 306/307 -> zeros
            }
        }
        __syncthreads();

        const unsigned short* bij = blane + ij * 5120;   // 10240 B per (i,j) group

        // ---------- 5 K=32 steps: (2 (k,l)-taps) x (16 channels); offsets fold ----------
        union F { uint4 u[2]; v16bf v; };
        #pragma unroll
        for (int q = 0; q < 5; ++q) {
            const int t0 = 2 * q, t1 = 2 * q + 1;      // t1==9 on q==4 -> zero slot

            // A fragments for both row groups; chunk1 = tap t0, chunk2 = tap t1
            F fa0, fa1;
            {
                const int dk = t0 / 3 - 1, dl = t0 % 3 - 1, off = (dk * S1 + dl) * AROW;
                bool v0 = ((unsigned)(d2r0 + dk) < DDIM) & ((unsigned)(d3r0 + dl) < DDIM);
                bool v1 = ((unsigned)(d2r1 + dk) < DDIM) & ((unsigned)(d3r1 + dl) < DDIM);
                fa0.u[0] = ((const uint4*)(v0 ? arow0 + off : azero))[0];
                fa1.u[0] = ((const uint4*)(v1 ? arow1 + off : azero))[0];
            }
            if (t1 < 9) {
                const int dk = t1 / 3 - 1, dl = t1 % 3 - 1, off = (dk * S1 + dl) * AROW;
                bool v0 = ((unsigned)(d2r0 + dk) < DDIM) & ((unsigned)(d3r0 + dl) < DDIM);
                bool v1 = ((unsigned)(d2r1 + dk) < DDIM) & ((unsigned)(d3r1 + dl) < DDIM);
                fa0.u[1] = ((const uint4*)(v0 ? arow0 + off : azero))[0];
                fa1.u[1] = ((const uint4*)(v1 ? arow1 + off : azero))[0];
            } else {
                fa0.u[1] = ((const uint4*)azero)[0];
                fa1.u[1] = ((const uint4*)azero)[0];
            }

            // B fragments: immediate-offset b128 loads from pre-expanded ws W
            F fb0, fb1;
            const uint4* pb0 = (const uint4*)(bij + q * 1024);        // frag 0
            const uint4* pb1 = (const uint4*)(bij + q * 1024 + 512);  // frag 1
            fb0.u[0] = pb0[0];  fb0.u[1] = pb0[1];
            fb1.u[0] = pb1[0];  fb1.u[1] = pb1[1];

            acc00 = __builtin_amdgcn_wmma_f32_16x16x32_bf16(
                        false, fa0.v, false, fb0.v, (short)0, acc00, false, false);
            acc01 = __builtin_amdgcn_wmma_f32_16x16x32_bf16(
                        false, fa0.v, false, fb1.v, (short)0, acc01, false, false);
            acc10 = __builtin_amdgcn_wmma_f32_16x16x32_bf16(
                        false, fa1.v, false, fb0.v, (short)0, acc10, false, false);
            acc11 = __builtin_amdgcn_wmma_f32_16x16x32_bf16(
                        false, fa1.v, false, fb1.v, (short)0, acc11, false, false);
        }
    }

    // ---------- store: lane holds 8 consecutive M rows for one N column ----------
    {
        union O { v8f v; float4 f[2]; };
        O o00, o01, o10, o11;
        o00.v = acc00; o01.v = acc01; o10.v = acc10; o11.v = acc11;
        int m0   = tile_m + wv * 32 + half * 8;      // row group 0
        int m1   = m0 + 16;                          // row group 1
        int sr0  = m0 - b * SPATIAL;
        int sr1  = m1 - b * SPATIAL;
        float* p00 = out + (size_t)(b * COUT + l15)      * SPATIAL + sr0;
        float* p01 = out + (size_t)(b * COUT + 16 + l15) * SPATIAL + sr0;
        float* p10 = out + (size_t)(b * COUT + l15)      * SPATIAL + sr1;
        float* p11 = out + (size_t)(b * COUT + 16 + l15) * SPATIAL + sr1;
        ((float4*)p00)[0] = o00.f[0];  ((float4*)p00)[1] = o00.f[1];
        ((float4*)p01)[0] = o01.f[0];  ((float4*)p01)[1] = o01.f[1];
        ((float4*)p10)[0] = o10.f[0];  ((float4*)p10)[1] = o10.f[1];
        ((float4*)p11)[0] = o11.f[0];  ((float4*)p11)[1] = o11.f[1];
    }
}

extern "C" void kernel_launch(void* const* d_in, const int* in_sizes, int n_in,
                              void* d_out, int out_size, void* d_ws, size_t ws_size,
                              hipStream_t stream) {
    const float* x  = (const float*)d_in[0];
    const float* Wt = (const float*)d_in[1];
    float* out = (float*)d_out;
    unsigned short* wb = (unsigned short*)d_ws;   // 92160 B of scratch

    prep_w_kernel<<<(WB_ELEMS + 255) / 256, 256, 0, stream>>>(Wt, wb);

    dim3 grid(MTOT / BM);   // 663552 / 256 = 2592 workgroups
    dim3 block(256);        // 8 wave32s per workgroup
    conv4d_wmma_kernel<<<grid, block, 0, stream>>>(x, wb, out);
}
